// StableLinearAttention_26173530701825
// MI455X (gfx1250) — matrix operand order repaired and checked by reading, not compiled, not run on hardware
//
#include <hip/hip_runtime.h>
#include <hip/hip_bf16.h>

// ---------------------------------------------------------------------------
// StableLinearAttention for MI455X (gfx1250, wave32, WMMA 16x16x32 f16).
//
//   qkv = x @ W_qkv                 -> q | k | v  (f16, custom layouts)
//   q   = softmax over d (64)       -> in-place on q buffer
//   ksm = softmax over n (8192)     -> folded into context GEMM A-fragments
//   ctx[b,h,d,e]   = sum_n ksm[n,d] v[n,e]          (WMMA, K = 8192)
//   Mcomb[b,hd,j]  = sum_e ctx[d,e] W_out[h64+e,j]  (WMMA, absorbs head mix)
//   out[b,n,j]     = q_sm @ Mcomb[b] + b_out        (WMMA, K = 512)
// ---------------------------------------------------------------------------

#define DIM      512
#define HEADS    8
#define DHEAD    64
#define BATCH    4
#define SEQ      8192
#define TOKENS   (BATCH * SEQ)      // 32768
#define QKV_COLS (3 * DIM)          // 1536

typedef __attribute__((ext_vector_type(16))) _Float16 v16h;
typedef __attribute__((ext_vector_type(8)))  _Float16 v8h;
typedef __attribute__((ext_vector_type(8)))  float    v8f;

// --------------------------- WMMA helpers ----------------------------------

__device__ __forceinline__ v8f wmma16(v16h a, v16h b, v8f c) {
    // D = A(16x32) * B(32x16) + C, f32 accumulate
    return __builtin_amdgcn_wmma_f32_16x16x32_f16(
        /*neg_a=*/false, a, /*neg_b=*/false, b,
        /*c_mod=*/(short)0, c, /*reuse_a=*/false, /*reuse_b=*/false);
}

// A-fragment (16x32, f16): lane (0..15) holds row M=lane, K = {0..7,16..23};
// lane (16..31) holds row M=lane-16, K = {8..15,24..31}.
// `base` is row-major [M x K] with leading dim `ld`, K contiguous.
__device__ __forceinline__ v16h load_frag_a(const _Float16* base, int ld,
                                            int row0, int k0) {
    int lane = threadIdx.x & 31;
    int r    = lane & 15;
    int kb   = k0 + ((lane >> 4) << 3);        // +8 for hi half-wave
    const _Float16* p = base + (size_t)(row0 + r) * ld + kb;
    v8h lo = *(const v8h*)(p);
    v8h hi = *(const v8h*)(p + 16);
    v16h a;
#pragma unroll
    for (int i = 0; i < 8; ++i) { a[i] = lo[i]; a[i + 8] = hi[i]; }
    return a;
}

// B-fragment (32x16, f16): lane (0..15) holds col N=lane, K = {0..7,16..23};
// lane (16..31) holds col N=lane-16, K = {8..15,24..31}.
// `baseT` is [N x K] (i.e. B transposed), K contiguous, leading dim `ld`.
__device__ __forceinline__ v16h load_frag_b(const _Float16* baseT, int ld,
                                            int col0, int k0) {
    int lane = threadIdx.x & 31;
    int c    = lane & 15;
    int kb   = k0 + ((lane >> 4) << 3);        // +8 for hi half-wave
    const _Float16* p = baseT + (size_t)(col0 + c) * ld + kb;
    v8h lo = *(const v8h*)(p);
    v8h hi = *(const v8h*)(p + 16);
    v16h b;
#pragma unroll
    for (int i = 0; i < 8; ++i) { b[i] = lo[i]; b[i + 8] = hi[i]; }
    return b;
}

// ------------------------- precision conversion ----------------------------

__global__ void cvt_f32_to_f16(const float* __restrict__ in,
                               _Float16* __restrict__ out, int n) {
    int i = blockIdx.x * 256 + threadIdx.x;
    if (i < n) out[i] = (_Float16)in[i];
}

// w: [rows x cols] f32 -> wt: [cols x rows] f16
__global__ void transpose_to_f16(const float* __restrict__ w,
                                 _Float16* __restrict__ wt, int rows, int cols) {
    int i = blockIdx.x * 256 + threadIdx.x;
    if (i < rows * cols) {
        int r = i / cols, c = i % cols;
        wt[(size_t)c * rows + r] = (_Float16)w[i];
    }
}

// ------------------------------ QKV GEMM -----------------------------------
// qkv = x @ W_qkv  (M=32768, N=1536, K=512). Each wave: 16x64 strip.
// Epilogue scatters q -> [token, h*64+d], k -> kT[bh][d][n], v -> vT[bh][e][n].
__global__ void qkv_gemm(const _Float16* __restrict__ xh,
                         const _Float16* __restrict__ wqkvT,
                         _Float16* __restrict__ q,
                         _Float16* __restrict__ kT,
                         _Float16* __restrict__ vT) {
    const int MT = TOKENS / 16;                       // 2048 row tiles
    int w   = blockIdx.x * (blockDim.x >> 5) + (threadIdx.x >> 5);
    int mt  = w % MT;
    int ns  = w / MT;                                 // 0..23 column strips
    int m0  = mt * 16;
    int col0 = ns * 64;

    v8f acc[4] = {v8f{}, v8f{}, v8f{}, v8f{}};
    for (int k0 = 0; k0 < DIM; k0 += 32) {
        if (k0 + 32 < DIM)
            __builtin_prefetch(xh + (size_t)(m0 + (threadIdx.x & 15)) * DIM + k0 + 32, 0, 1);
        v16h a = load_frag_a(xh, DIM, m0, k0);
#pragma unroll
        for (int t = 0; t < 4; ++t) {
            v16h b = load_frag_b(wqkvT, DIM, col0 + 16 * t, k0);
            acc[t] = wmma16(a, b, acc[t]);
        }
    }

    int lane = threadIdx.x & 31;
    int hi = lane >> 4, cc = lane & 15;
#pragma unroll
    for (int t = 0; t < 4; ++t) {
        int col = col0 + 16 * t + cc;
#pragma unroll
        for (int r = 0; r < 8; ++r) {
            int row = m0 + r + 8 * hi;                // token index
            _Float16 val = (_Float16)acc[t][r];
            if (col < DIM) {                          // q region
                q[(size_t)row * DIM + col] = val;
            } else if (col < 2 * DIM) {               // k region -> kT[bh][d][n]
                int c2 = col - DIM, h = c2 >> 6, d = c2 & 63;
                int b = row >> 13, n = row & (SEQ - 1);
                kT[(((size_t)(b * HEADS + h)) * DHEAD + d) * SEQ + n] = val;
            } else {                                  // v region -> vT[bh][e][n]
                int c2 = col - 2 * DIM, h = c2 >> 6, e = c2 & 63;
                int b = row >> 13, n = row & (SEQ - 1);
                vT[(((size_t)(b * HEADS + h)) * DHEAD + e) * SEQ + n] = val;
            }
        }
    }
}

// -------------------------- q softmax over d (64) --------------------------
// One wave per (token, head); lane handles 2 of 64 values. In-place.
__global__ void softmax_q(_Float16* __restrict__ q) {
    int w     = blockIdx.x * (blockDim.x >> 5) + (threadIdx.x >> 5);
    int lane  = threadIdx.x & 31;
    int token = w >> 3;
    int h     = w & 7;
    _Float16* p = q + (size_t)token * DIM + h * DHEAD;
    float v0 = (float)p[lane], v1 = (float)p[lane + 32];
    float m = fmaxf(v0, v1);
#pragma unroll
    for (int off = 16; off; off >>= 1) m = fmaxf(m, __shfl_xor(m, off, 32));
    float e0 = __expf(v0 - m), e1 = __expf(v1 - m);
    float s = e0 + e1;
#pragma unroll
    for (int off = 16; off; off >>= 1) s += __shfl_xor(s, off, 32);
    float inv = 1.0f / s;
    p[lane]      = (_Float16)(e0 * inv);
    p[lane + 32] = (_Float16)(e1 * inv);
}

// ------------------- k column stats: max & sum-exp over n ------------------
// One block per (b,h). Thread t: column d = t/4, quarter sub = t%4.
__global__ void k_stats(const _Float16* __restrict__ kT,
                        float* __restrict__ kmax, float* __restrict__ ksum) {
    int bh = blockIdx.x, t = threadIdx.x;
    int d = t >> 2, sub = t & 3;
    const _Float16* p = kT + ((size_t)bh * DHEAD + d) * SEQ + sub * (SEQ / 4);
    __shared__ float red[256];

    float m = -3.0e38f;
    for (int i = 0; i < SEQ / 4; ++i) m = fmaxf(m, (float)p[i]);
    red[t] = m;
    __syncthreads();
    if (sub == 0) {
        m = fmaxf(fmaxf(red[t], red[t + 1]), fmaxf(red[t + 2], red[t + 3]));
        red[t] = m;
    }
    __syncthreads();
    m = red[d * 4];
    __syncthreads();

    float s = 0.0f;
    for (int i = 0; i < SEQ / 4; ++i) s += __expf((float)p[i] - m);
    red[t] = s;
    __syncthreads();
    if (sub == 0) {
        s = red[t] + red[t + 1] + red[t + 2] + red[t + 3];
        kmax[bh * DHEAD + d] = m;
        ksum[bh * DHEAD + d] = s;
    }
}

// ---------------------- context GEMM (softmax_k fused) ---------------------
// ctx[bh][d][e] = sum_n exp(k[n,d]-max[d])/sum[d] * v[n,e].  One block per bh.
// Wave w: d-tile = w&3, e-tiles = {2*(w>>2), 2*(w>>2)+1}; K = n = 8192.
__global__ void context_gemm(const _Float16* __restrict__ kT,
                             const _Float16* __restrict__ vT,
                             const float* __restrict__ kmax,
                             const float* __restrict__ ksum,
                             _Float16* __restrict__ ctx) {
    int bh   = blockIdx.x;
    int w    = threadIdx.x >> 5;
    int lane = threadIdx.x & 31;
    int dt   = w & 3;
    int et0  = (w >> 2) * 2;

    int d = dt * 16 + (lane & 15);                    // A-fragment lane row
    float mx = kmax[bh * DHEAD + d];
    float rs = 1.0f / ksum[bh * DHEAD + d];

    const _Float16* kbase = kT + (size_t)bh * DHEAD * SEQ;
    const _Float16* vbase = vT + (size_t)bh * DHEAD * SEQ;

    v8f acc0 = {}, acc1 = {};
    for (int k0 = 0; k0 < SEQ; k0 += 32) {
        v16h a = load_frag_a(kbase, SEQ, dt * 16, k0);
#pragma unroll
        for (int i = 0; i < 16; ++i)                  // fused k-softmax
            a[i] = (_Float16)(__expf((float)a[i] - mx) * rs);
        v16h b0 = load_frag_b(vbase, SEQ, et0 * 16, k0);
        v16h b1 = load_frag_b(vbase, SEQ, et0 * 16 + 16, k0);
        acc0 = wmma16(a, b0, acc0);
        acc1 = wmma16(a, b1, acc1);
    }

    int hi = lane >> 4, cc = lane & 15;
#pragma unroll
    for (int r = 0; r < 8; ++r) {
        int row = dt * 16 + r + 8 * hi;               // d
        ctx[((size_t)bh * DHEAD + row) * DHEAD + et0 * 16 + cc]      = (_Float16)acc0[r];
        ctx[((size_t)bh * DHEAD + row) * DHEAD + et0 * 16 + 16 + cc] = (_Float16)acc1[r];
    }
}

// ----------------- Mcomb GEMM: absorb W_out into context -------------------
// McombT[b][j][h*64+d] = sum_e ctx[bh][d][e] * W_out[h*64+e][j]
__global__ void mcomb_gemm(const _Float16* __restrict__ ctx,
                           const _Float16* __restrict__ woutT,
                           _Float16* __restrict__ mcT) {
    int w  = blockIdx.x * (blockDim.x >> 5) + (threadIdx.x >> 5);
    int jt = w & 31;                                  // j tile (512/16)
    int dt = (w >> 5) & 3;                            // d tile
    int h  = (w >> 7) & 7;
    int b  = w >> 10;
    int bh = b * HEADS + h;

    v8f acc = {};
#pragma unroll
    for (int k0 = 0; k0 < DHEAD; k0 += 32) {
        v16h a  = load_frag_a(ctx + (size_t)bh * DHEAD * DHEAD, DHEAD, dt * 16, k0);
        v16h bb = load_frag_b(woutT, DIM, jt * 16, h * DHEAD + k0);
        acc = wmma16(a, bb, acc);
    }

    int lane = threadIdx.x & 31;
    int hi = lane >> 4, cc = lane & 15;
    int j = jt * 16 + cc;
#pragma unroll
    for (int r = 0; r < 8; ++r) {
        int hd = h * DHEAD + dt * 16 + r + 8 * hi;
        mcT[((size_t)b * DIM + j) * DIM + hd] = (_Float16)acc[r];
    }
}

// --------------------------- final output GEMM -----------------------------
// out[token][j] = q_sm[token] . McombT[b] + b_out[j]   (K = 512, f32 out)
__global__ void out_gemm(const _Float16* __restrict__ q,
                         const _Float16* __restrict__ mcT,
                         const float* __restrict__ bias,
                         float* __restrict__ out) {
    const int MT = TOKENS / 16;                       // 2048
    int w   = blockIdx.x * (blockDim.x >> 5) + (threadIdx.x >> 5);
    int mt  = w % MT;
    int ns  = w / MT;                                 // 0..7
    int m0  = mt * 16;
    int col0 = ns * 64;
    int b   = m0 >> 13;
    const _Float16* mc = mcT + (size_t)b * DIM * DIM;

    v8f acc[4] = {v8f{}, v8f{}, v8f{}, v8f{}};
    for (int k0 = 0; k0 < DIM; k0 += 32) {
        if (k0 + 32 < DIM)
            __builtin_prefetch(q + (size_t)(m0 + (threadIdx.x & 15)) * DIM + k0 + 32, 0, 1);
        v16h a = load_frag_a(q, DIM, m0, k0);
#pragma unroll
        for (int t = 0; t < 4; ++t) {
            v16h bb = load_frag_b(mc, DIM, col0 + 16 * t, k0);
            acc[t] = wmma16(a, bb, acc[t]);
        }
    }

    int lane = threadIdx.x & 31;
    int hi = lane >> 4, cc = lane & 15;
#pragma unroll
    for (int t = 0; t < 4; ++t) {
        int col = col0 + 16 * t + cc;
        float bj = bias[col];
#pragma unroll
        for (int r = 0; r < 8; ++r)
            out[(size_t)(m0 + r + 8 * hi) * DIM + col] = acc[t][r] + bj;
    }
}

// ------------------------------- launcher ----------------------------------

extern "C" void kernel_launch(void* const* d_in, const int* in_sizes, int n_in,
                              void* d_out, int out_size, void* d_ws, size_t ws_size,
                              hipStream_t stream) {
    const float* x     = (const float*)d_in[0];   // [4, 8192, 512]
    const float* W_qkv = (const float*)d_in[1];   // [512, 1536]
    const float* W_out = (const float*)d_in[2];   // [512, 512]
    const float* b_out = (const float*)d_in[3];   // [512]
    float* out = (float*)d_out;                   // [4, 8192, 512]

    char* ws = (char*)d_ws;
    size_t off = 0;
    auto take = [&](size_t bytes) { char* p = ws + off; off += (bytes + 255) & ~(size_t)255; return p; };

    _Float16* xh     = (_Float16*)take((size_t)TOKENS * DIM * 2);        // 32 MB
    _Float16* wqkvT  = (_Float16*)take((size_t)QKV_COLS * DIM * 2);      // 1.5 MB
    _Float16* woutT  = (_Float16*)take((size_t)DIM * DIM * 2);           // 0.5 MB
    _Float16* q      = (_Float16*)take((size_t)TOKENS * DIM * 2);        // 32 MB
    _Float16* kT     = (_Float16*)take((size_t)BATCH * HEADS * DHEAD * SEQ * 2); // 32 MB
    _Float16* vT     = (_Float16*)take((size_t)BATCH * HEADS * DHEAD * SEQ * 2); // 32 MB
    float*    kmax   = (float*)take((size_t)BATCH * HEADS * DHEAD * 4);
    float*    ksum   = (float*)take((size_t)BATCH * HEADS * DHEAD * 4);
    _Float16* ctx    = (_Float16*)take((size_t)BATCH * HEADS * DHEAD * DHEAD * 2);
    _Float16* mcT    = (_Float16*)take((size_t)BATCH * DIM * DIM * 2);   // 2 MB

    const dim3 blk(256);

    // 1) precision conversion / weight transposes
    cvt_f32_to_f16<<<(TOKENS * DIM + 255) / 256, blk, 0, stream>>>(x, xh, TOKENS * DIM);
    transpose_to_f16<<<(DIM * QKV_COLS + 255) / 256, blk, 0, stream>>>(W_qkv, wqkvT, DIM, QKV_COLS);
    transpose_to_f16<<<(DIM * DIM + 255) / 256, blk, 0, stream>>>(W_out, woutT, DIM, DIM);

    // 2) QKV GEMM: (32768/16) * (1536/64) = 49152 waves, 8 waves/block
    qkv_gemm<<<(TOKENS / 16) * (QKV_COLS / 64) / 8, blk, 0, stream>>>(xh, wqkvT, q, kT, vT);

    // 3) q softmax over d: 32768*8 waves
    softmax_q<<<TOKENS * HEADS / 8, blk, 0, stream>>>(q);

    // 4) k column stats: one block per (b,h)
    k_stats<<<BATCH * HEADS, blk, 0, stream>>>(kT, kmax, ksum);

    // 5) context GEMM with fused k-softmax: one block per (b,h)
    context_gemm<<<BATCH * HEADS, blk, 0, stream>>>(kT, vT, kmax, ksum, ctx);

    // 6) Mcomb GEMM: 4*8*4*32 = 4096 waves
    mcomb_gemm<<<BATCH * HEADS * 4 * 32 / 8, blk, 0, stream>>>(ctx, woutT, mcT);

    // 7) final GEMM + bias: (32768/16) * (512/64) = 16384 waves
    out_gemm<<<(TOKENS / 16) * (DIM / 64) / 8, blk, 0, stream>>>(q, mcT, b_out, out);
}
